// Model_23871428231852
// MI455X (gfx1250) — matrix operand order repaired and checked by reading, not compiled, not run on hardware
//
#include <hip/hip_runtime.h>
#include <hip/hip_bf16.h>
#include <math.h>

// ---------------- CDNA5 WMMA types ----------------
typedef __attribute__((ext_vector_type(16))) __bf16 v16bf;
typedef __attribute__((ext_vector_type(8)))  float  v8f;

union BF16Frag { uint4 u4[2]; v16bf v; };

static __device__ __forceinline__ v8f wmma_bf16(v16bf a, v16bf b, v8f c) {
  // (neg_a, A, neg_b, B, c_mod, C, reuse_a, reuse_b)
  return __builtin_amdgcn_wmma_f32_16x16x32_bf16(false, a, false, b, (short)0, c,
                                                 false, false);
}

// CDNA5 async global->LDS copy (ASYNCcnt-tracked), 16B per lane.
// vdst VGPR holds the LDS byte offset (low 32 bits of a generic shared ptr),
// vaddr pair holds the 64-bit global address (GV mode).
static __device__ __forceinline__ void async_b128(const void* g, void* l) {
  unsigned l32 = (unsigned)(uintptr_t)l;
  unsigned long long g64 = (unsigned long long)(uintptr_t)g;
  asm volatile("global_load_async_to_lds_b128 %0, %1, off"
               :: "v"(l32), "v"(g64)
               : "memory");
}
static __device__ __forceinline__ void async_wait0() {
  asm volatile("s_wait_asynccnt 0x0" ::: "memory");
}
static __device__ __forceinline__ void async_wait2() {
  asm volatile("s_wait_asynccnt 0x2" ::: "memory");
}
static __device__ __forceinline__ void async_wait4() {
  asm volatile("s_wait_asynccnt 0x4" ::: "memory");
}

// ---------------- problem constants ----------------
#define BB   4
#define TT   2048
#define VV   50257
#define EE   1024
#define HH   16
#define DD   64
#define E3   3072
#define E4   4096
#define NTOK 8192   // B*T

// =====================================================================
// 1) Embedding: x[bt,e] = wte[ids[bt],e] + wpe[t,e]
// =====================================================================
__global__ __launch_bounds__(256) void embed_kernel(const int* __restrict__ ids,
                                                    const float* __restrict__ wte,
                                                    const float* __restrict__ wpe,
                                                    float* __restrict__ x) {
  size_t idx = (size_t)blockIdx.x * 256 + threadIdx.x;   // NTOK*EE total
  int bt = (int)(idx >> 10);
  int e  = (int)(idx & 1023);
  int t  = bt & (TT - 1);
  x[idx] = wte[(size_t)ids[bt] * EE + e] + wpe[(size_t)t * EE + e];
}

// =====================================================================
// 2) LayerNorm over E=1024, output bf16. One block (256 thr) per row.
// =====================================================================
__global__ __launch_bounds__(256) void ln_kernel(const float* __restrict__ x,
                                                 const float* __restrict__ g,
                                                 const float* __restrict__ b,
                                                 __bf16* __restrict__ out) {
  __shared__ float red[256];
  const int tid = threadIdx.x;
  const size_t row = blockIdx.x;
  const float* xr = x + row * EE;

  float vals[4];
  float s = 0.f;
#pragma unroll
  for (int i = 0; i < 4; ++i) { vals[i] = xr[tid + i * 256]; s += vals[i]; }
  red[tid] = s; __syncthreads();
  for (int o = 128; o > 0; o >>= 1) { if (tid < o) red[tid] += red[tid + o]; __syncthreads(); }
  const float mu = red[0] * (1.f / EE);
  __syncthreads();

  float s2 = 0.f;
#pragma unroll
  for (int i = 0; i < 4; ++i) { float d = vals[i] - mu; s2 += d * d; }
  red[tid] = s2; __syncthreads();
  for (int o = 128; o > 0; o >>= 1) { if (tid < o) red[tid] += red[tid + o]; __syncthreads(); }
  const float rinv = rsqrtf(red[0] * (1.f / EE) + 1e-5f);

#pragma unroll
  for (int i = 0; i < 4; ++i) {
    int e = tid + i * 256;
    out[row * EE + e] = (__bf16)((vals[i] - mu) * rinv * g[e] + b[e]);
  }
}

// =====================================================================
// 3) Weight conversion / packing (all weights stored TRANSPOSED [N][K]
//    so GEMM LDS staging is pure contiguous b128 async copies)
// =====================================================================
// in: f32 [K][N] row-major -> outT: bf16 [N][K]
__global__ __launch_bounds__(256) void cvt_bf16_t_kernel(const float* __restrict__ in,
                                                         __bf16* __restrict__ outT,
                                                         int K, int N) {
  size_t i = (size_t)blockIdx.x * 256 + threadIdx.x;   // N*K total
  if (i >= (size_t)K * N) return;
  int n = (int)(i / K);
  int k = (int)(i % K);
  outT[i] = (__bf16)in[(size_t)k * N + n];
}

// wq/wk/wv are (H,E,D); pack TRANSPOSED: outT [3E][E], row j = (m,h,d), col c.
__global__ __launch_bounds__(256) void pack_qkv_t_kernel(const float* __restrict__ wq,
                                                         const float* __restrict__ wk,
                                                         const float* __restrict__ wv,
                                                         __bf16* __restrict__ outT) {
  size_t i = (size_t)blockIdx.x * 256 + threadIdx.x;   // E3*EE total
  if (i >= (size_t)E3 * EE) return;
  int j  = (int)(i / EE);
  int c  = (int)(i % EE);
  int m  = j >> 10;
  int jj = j & 1023;
  const float* w = (m == 0) ? wq : (m == 1) ? wk : wv;
  outT[i] = (__bf16)w[(size_t)(jj >> 6) * (EE * DD) + (size_t)c * DD + (jj & 63)];
}

__global__ __launch_bounds__(256) void pack_bias_kernel(const float* __restrict__ bq,
                                                        const float* __restrict__ bk,
                                                        const float* __restrict__ bv,
                                                        float* __restrict__ out) {
  int i = blockIdx.x * 256 + threadIdx.x;
  if (i >= E3) return;
  out[i] = (i < 1024) ? bq[i] : (i < 2048) ? bk[i - 1024] : bv[i - 2048];
}

// =====================================================================
// 4) Tiled WMMA GEMM: C[M,N] = A[M,K](bf16) * B[K,N](bf16, given as B^T [N][K])
//    Block tile 128x128, 8 waves (4x2), wave tile 32x64 => 8 WMMA per k-step.
//    Double-buffered async global->LDS pipeline (4 async ops / stage / thread).
//    Compile-time specialized epilogue: +bias, exact GELU, residual add,
//    bf16 / f32 output, block-uniform V-split into [B,H,D,T].
// =====================================================================
template <bool GELU, bool RES, bool OUTB, bool SPLITV, bool OUTF>
__global__ __launch_bounds__(256) void gemm_bf16_kernel(
    const __bf16* __restrict__ A, const __bf16* __restrict__ Bt,
    const float* __restrict__ bias, const float* __restrict__ res,
    __bf16* __restrict__ outB, __bf16* __restrict__ vt,
    float* __restrict__ outF,
    int M, int N, int K) {
  __shared__ __bf16 As[2][128][40];   // [buf][m][k], rows 80B (16B aligned)
  __shared__ __bf16 Bs[2][128][40];   // [buf][n][k]

  const int tid = threadIdx.x;
  const int wave = tid >> 5, lane = tid & 31;
  const int wm = wave & 3, wn = wave >> 2;       // 4 x 2 wave grid
  const int lhalf = lane >> 4, lmod = lane & 15;
  const int m0 = blockIdx.x * 128, n0 = blockIdx.y * 128;

  const int nsteps = K >> 5;

  auto stage = [&](int s, int b) {
    const int k0 = s << 5;
#pragma unroll
    for (int p = 0; p < 2; ++p) {
      int uid = p * 256 + tid;
      int row = uid >> 2, c0 = (uid & 3) * 8;
      async_b128(A + (size_t)(m0 + row) * K + k0 + c0, &As[b][row][c0]);
    }
#pragma unroll
    for (int p = 0; p < 2; ++p) {
      int uid = p * 256 + tid;
      int nn = uid >> 2, c0 = (uid & 3) * 8;
      async_b128(Bt + (size_t)(n0 + nn) * K + k0 + c0, &Bs[b][nn][c0]);
    }
  };

  v8f acc[2][4];
#pragma unroll
  for (int mi = 0; mi < 2; ++mi)
#pragma unroll
    for (int ni = 0; ni < 4; ++ni) acc[mi][ni] = (v8f)0.f;

  stage(0, 0);
  if (nsteps > 1) stage(1, 1);

  for (int i = 0; i < nsteps; ++i) {
    if (i + 1 < nsteps) async_wait4();   // stage i done, stage i+1 may fly
    else                async_wait0();
    __syncthreads();
    const int b = i & 1;

    BF16Frag af[2], bf[4];
#pragma unroll
    for (int mi = 0; mi < 2; ++mi) {
      int row = wm * 32 + mi * 16 + lmod;          // A: row = lane%16
      af[mi].u4[0] = *(const uint4*)&As[b][row][lhalf * 8];
      af[mi].u4[1] = *(const uint4*)&As[b][row][16 + lhalf * 8];
    }
#pragma unroll
    for (int ni = 0; ni < 4; ++ni) {
      int nn = wn * 64 + ni * 16 + lmod;           // B: col = lane%16
      bf[ni].u4[0] = *(const uint4*)&Bs[b][nn][lhalf * 8];
      bf[ni].u4[1] = *(const uint4*)&Bs[b][nn][16 + lhalf * 8];
    }
#pragma unroll
    for (int mi = 0; mi < 2; ++mi)
#pragma unroll
      for (int ni = 0; ni < 4; ++ni)
        acc[mi][ni] = wmma_bf16(af[mi].v, bf[ni].v, acc[mi][ni]);

    __syncthreads();                      // all waves done reading buf b
    if (i + 2 < nsteps) stage(i + 2, b);  // overwrite buf b with stage i+2
  }

  // --- epilogue: C layout row = lhalf*8 + r, col = lane%16 ---
  // V-region boundary (2048) is a multiple of the 128-wide block tile, so
  // the split test is block-uniform.
  const bool isV = SPLITV && (n0 >= 2048);
#pragma unroll
  for (int mi = 0; mi < 2; ++mi)
#pragma unroll
    for (int ni = 0; ni < 4; ++ni)
#pragma unroll
      for (int r = 0; r < 8; ++r) {
        int row = m0 + wm * 32 + mi * 16 + lhalf * 8 + r;
        int col = n0 + wn * 64 + ni * 16 + lmod;
        float v = acc[mi][ni][r] + bias[col];
        if (GELU) v = 0.5f * v * (1.f + erff(v * 0.70710678118f));
        if (RES)  v += res[(size_t)row * N + col];
        if (OUTB) {
          if (SPLITV && isV) {
            // V head data -> transposed [B*H*D, T] for scatter-free attention
            int hd = col - 2048;                  // h*64 + d
            int bb = row >> 11, t = row & (TT - 1);
            vt[((size_t)bb * 1024 + hd) * TT + t] = (__bf16)v;
          } else {
            outB[(size_t)row * N + col] = (__bf16)v;
          }
        }
        if (OUTF) outF[(size_t)row * N + col] = v;
      }
}

// =====================================================================
// 5) Flash attention. Grid: (B*H, T/128). Block 256 = 8 waves; each wave
//    owns 16 query rows. Double-buffered async K/V streaming (32-key tiles)
//    with online softmax. Q,K from qkv [B*T,3E]; V from vt [B*H*D, T].
//    Key loop is split: fully-visible tiles (no masking) + 4 diagonal tiles.
//    Q is pre-scaled by 1/sqrt(D)=0.125 (exact in bf16).
// =====================================================================
__global__ __launch_bounds__(256) void attn_kernel(const __bf16* __restrict__ qkv,
                                                   const __bf16* __restrict__ vt,
                                                   __bf16* __restrict__ o) {
  __shared__ __bf16 Kt[2][32][72];     // [buf][key][d], rows 144B
  __shared__ __bf16 Vt[2][64][40];     // [buf][d][key], rows 80B
  __shared__ __bf16 Pb[8][16][40];     // per-wave P staging [qrow][key]

  const int bh = blockIdx.x;
  const int b = bh / HH, h = bh % HH;
  const int qtile = blockIdx.y;
  const int tid = threadIdx.x, wave = tid >> 5, lane = tid & 31;
  const int lhalf = lane >> 4, lmod = lane & 15;

  const int qrow0 = qtile * 128 + wave * 16;
  const __bf16* kbase = qkv + 1024 + h * DD;          // K columns
  const __bf16* vbase = vt + ((size_t)b * 1024 + h * DD) * TT;

  auto stageKV = [&](int it, int bf) {
    const int s0 = it << 5;
    {  // K tile [32 keys][64 d] : 1 x b128 per thread
      int s = tid >> 3, d0 = (tid & 7) * 8;
      async_b128(kbase + (size_t)(b * TT + s0 + s) * E3 + d0, &Kt[bf][s][d0]);
    }
    {  // V tile [64 d][32 keys] : 1 x b128 per thread (already transposed)
      int d = tid >> 2, c0 = (tid & 3) * 8;
      async_b128(vbase + (size_t)d * TT + s0 + c0, &Vt[bf][d][c0]);
    }
  };

  // ---- load Q fragments (A layout, 16x32 per 32-wide d-chunk) ----
  // Pre-scaled by 1/sqrt(D) = 0.125 (power of two: exact in bf16).
  const __bf16* qptr = qkv + (size_t)(b * TT + qrow0 + lmod) * E3 + h * DD;
  BF16Frag qf[2];
#pragma unroll
  for (int kc = 0; kc < 2; ++kc) {
    qf[kc].u4[0] = *(const uint4*)(qptr + kc * 32 + lhalf * 8);
    qf[kc].u4[1] = *(const uint4*)(qptr + kc * 32 + 16 + lhalf * 8);
    qf[kc].v = qf[kc].v * (__bf16)0.125f;
  }

  v8f Oacc[4];
#pragma unroll
  for (int j = 0; j < 4; ++j) Oacc[j] = (v8f)0.f;
  float mrow[8], ssum[8];
#pragma unroll
  for (int r = 0; r < 8; ++r) { mrow[r] = -1e30f; ssum[r] = 0.f; }

  const int nmain = qtile * 4;                  // fully causally-visible tiles
  const int niter = nmain + 4;                  // + 4 diagonal tiles
  const int trow_base = qrow0 + lhalf * 8;      // C-layout rows for this lane

  stageKV(0, 0);
  if (niter > 1) stageKV(1, 1);

  // One key tile; `masked` is a literal at each call site -> folds away.
  auto tile_body = [&](int it, bool masked) {
    if (it + 1 < niter) async_wait2();
    else                async_wait0();
    __syncthreads();
    const int bf = it & 1;
    const int s0 = it << 5;

    // ---- S = Q * K^T : two 16-col groups, contraction d in 2 chunks ----
    v8f S[2];
#pragma unroll
    for (int j = 0; j < 2; ++j) {
      v8f a = (v8f)0.f;
#pragma unroll
      for (int kc = 0; kc < 2; ++kc) {
        BF16Frag kf;
        kf.u4[0] = *(const uint4*)&Kt[bf][j * 16 + lmod][kc * 32 + lhalf * 8];
        kf.u4[1] = *(const uint4*)&Kt[bf][j * 16 + lmod][kc * 32 + 16 + lhalf * 8];
        a = wmma_bf16(qf[kc].v, kf.v, a);
      }
      S[j] = a;
    }

    // ---- (diagonal only) causal mask, then row max ----
    float rmax[8];
#pragma unroll
    for (int r = 0; r < 8; ++r) {
      if (masked) {
        int tg = trow_base + r;
#pragma unroll
        for (int j = 0; j < 2; ++j) {
          int s = s0 + j * 16 + lmod;
          if (s > tg) S[j][r] = -1e30f;
        }
      }
      rmax[r] = fmaxf(S[0][r], S[1][r]);
    }
#pragma unroll
    for (int off = 1; off < 16; off <<= 1)
#pragma unroll
      for (int r = 0; r < 8; ++r)
        rmax[r] = fmaxf(rmax[r], __shfl_xor(rmax[r], off, 32));

    // ---- online softmax update ----
    float corr[8];
#pragma unroll
    for (int r = 0; r < 8; ++r) {
      float nm = fmaxf(mrow[r], rmax[r]);
      corr[r] = __expf(mrow[r] - nm);
      mrow[r] = nm;
    }
#pragma unroll
    for (int j = 0; j < 4; ++j)
#pragma unroll
      for (int r = 0; r < 8; ++r) Oacc[j][r] *= corr[r];

    float rsum[8];
#pragma unroll
    for (int r = 0; r < 8; ++r) {
      float p0 = __expf(S[0][r] - mrow[r]);
      float p1 = __expf(S[1][r] - mrow[r]);
      S[0][r] = p0; S[1][r] = p1;
      rsum[r] = p0 + p1;
    }
#pragma unroll
    for (int off = 1; off < 16; off <<= 1)
#pragma unroll
      for (int r = 0; r < 8; ++r) rsum[r] += __shfl_xor(rsum[r], off, 32);
#pragma unroll
    for (int r = 0; r < 8; ++r) ssum[r] = ssum[r] * corr[r] + rsum[r];

    // ---- P: C-layout -> A-layout via per-wave LDS staging ----
#pragma unroll
    for (int j = 0; j < 2; ++j)
#pragma unroll
      for (int r = 0; r < 8; ++r)
        Pb[wave][lhalf * 8 + r][j * 16 + lmod] = (__bf16)S[j][r];

    BF16Frag pf;
    pf.u4[0] = *(const uint4*)&Pb[wave][lmod][lhalf * 8];
    pf.u4[1] = *(const uint4*)&Pb[wave][lmod][16 + lhalf * 8];

    // ---- O += P * V : four 16-col output groups, contraction = 32 keys ----
#pragma unroll
    for (int j = 0; j < 4; ++j) {
      BF16Frag vf;
      vf.u4[0] = *(const uint4*)&Vt[bf][j * 16 + lmod][lhalf * 8];
      vf.u4[1] = *(const uint4*)&Vt[bf][j * 16 + lmod][16 + lhalf * 8];
      Oacc[j] = wmma_bf16(pf.v, vf.v, Oacc[j]);
    }

    __syncthreads();                          // all waves done with buf bf
    if (it + 2 < niter) stageKV(it + 2, bf);  // prefetch into buf bf
  };

  for (int it = 0; it < nmain; ++it) tile_body(it, false);   // mask-free
  for (int it = nmain; it < niter; ++it) tile_body(it, true); // diagonal

  // ---- normalize and write out (heads concatenated) ----
  float inv[8];
#pragma unroll
  for (int r = 0; r < 8; ++r) inv[r] = 1.0f / ssum[r];
#pragma unroll
  for (int j = 0; j < 4; ++j)
#pragma unroll
    for (int r = 0; r < 8; ++r) {
      int row = trow_base + r;
      int col = h * DD + j * 16 + lmod;
      o[(size_t)(b * TT + row) * EE + col] = (__bf16)(Oacc[j][r] * inv[r]);
    }
}

// =====================================================================
// Host-side orchestration
// =====================================================================
extern "C" void kernel_launch(void* const* d_in, const int* in_sizes, int n_in,
                              void* d_out, int out_size, void* d_ws, size_t ws_size,
                              hipStream_t stream) {
  (void)in_sizes; (void)n_in; (void)out_size; (void)ws_size;

  const int*   ids    = (const int*)  d_in[0];
  const float* wte    = (const float*)d_in[1];
  const float* wpe    = (const float*)d_in[2];
  const float* wq     = (const float*)d_in[3];
  const float* wk     = (const float*)d_in[4];
  const float* wv     = (const float*)d_in[5];
  const float* bq     = (const float*)d_in[6];
  const float* bk     = (const float*)d_in[7];
  const float* bv     = (const float*)d_in[8];
  const float* wproj  = (const float*)d_in[9];
  const float* bproj  = (const float*)d_in[10];
  const float* ln1_g  = (const float*)d_in[11];
  const float* ln1_b  = (const float*)d_in[12];
  const float* ln2_g  = (const float*)d_in[13];
  const float* ln2_b  = (const float*)d_in[14];
  const float* w_fc   = (const float*)d_in[15];
  const float* b_fc   = (const float*)d_in[16];
  const float* w_out  = (const float*)d_in[17];
  const float* b_out  = (const float*)d_in[18];
  float* out = (float*)d_out;

  // ---- workspace layout (bytes). m1 aliases the dead qkv+o region. ----
  char* ws = (char*)d_ws;
  const size_t o_x    = 0;                                   // f32 [8192,1024]
  const size_t o_x2   = o_x   + (size_t)NTOK * EE * 4;       // f32 [8192,1024]
  const size_t o_h    = o_x2  + (size_t)NTOK * EE * 4;       // bf16 [8192,1024] (h1 & h2)
  const size_t o_qkv  = o_h   + (size_t)NTOK * EE * 2;       // bf16 [8192,3072]
  const size_t o_o    = o_qkv + (size_t)NTOK * E3 * 2;       // bf16 [8192,1024]
  const size_t o_m1   = o_qkv;                               // bf16 [8192,4096] (alias)
  const size_t o_vt   = o_o   + (size_t)NTOK * EE * 2;       // bf16 [B*H*D, T]
  const size_t o_wqkv = o_vt  + (size_t)BB * EE * TT * 2;    // bf16 [3072,1024] (T)
  const size_t o_wpj  = o_wqkv+ (size_t)EE * E3 * 2;         // bf16 [1024,1024] (T)
  const size_t o_wfc  = o_wpj + (size_t)EE * EE * 2;         // bf16 [4096,1024] (T)
  const size_t o_wo   = o_wfc + (size_t)EE * E4 * 2;         // bf16 [1024,4096] (T)
  const size_t o_bqkv = o_wo  + (size_t)E4 * EE * 2;         // f32 [3072]

  float*  x      = (float*) (ws + o_x);
  float*  x2     = (float*) (ws + o_x2);
  __bf16* hbuf   = (__bf16*)(ws + o_h);
  __bf16* qkv    = (__bf16*)(ws + o_qkv);
  __bf16* obuf   = (__bf16*)(ws + o_o);
  __bf16* m1     = (__bf16*)(ws + o_m1);
  __bf16* vtbuf  = (__bf16*)(ws + o_vt);
  __bf16* wqkv_t = (__bf16*)(ws + o_wqkv);
  __bf16* wpj_t  = (__bf16*)(ws + o_wpj);
  __bf16* wfc_t  = (__bf16*)(ws + o_wfc);
  __bf16* wo_t   = (__bf16*)(ws + o_wo);
  float*  bqkv   = (float*) (ws + o_bqkv);

  // ---- weight prep (all transposed to [N][K] bf16) ----
  cvt_bf16_t_kernel<<<((size_t)EE * EE + 255) / 256, 256, 0, stream>>>(wproj, wpj_t, EE, EE);
  cvt_bf16_t_kernel<<<((size_t)EE * E4 + 255) / 256, 256, 0, stream>>>(w_fc, wfc_t, EE, E4);
  cvt_bf16_t_kernel<<<((size_t)E4 * EE + 255) / 256, 256, 0, stream>>>(w_out, wo_t, E4, EE);
  pack_qkv_t_kernel<<<((size_t)E3 * EE + 255) / 256, 256, 0, stream>>>(wq, wk, wv, wqkv_t);
  pack_bias_kernel<<<(E3 + 255) / 256, 256, 0, stream>>>(bq, bk, bv, bqkv);

  // ---- forward ----
  embed_kernel<<<((size_t)NTOK * EE) / 256, 256, 0, stream>>>(ids, wte, wpe, x);
  ln_kernel<<<NTOK, 256, 0, stream>>>(x, ln1_g, ln1_b, hbuf);

  // QKV: [8192,1024] x [1024,3072] -> qkv bf16 (+bias); V split to vt [B,H,D,T]
  gemm_bf16_kernel<false, false, true, true, false>
      <<<dim3(NTOK / 128, E3 / 128), 256, 0, stream>>>(
      hbuf, wqkv_t, bqkv, nullptr, qkv, vtbuf, nullptr, NTOK, E3, EE);

  // causal multi-head flash attention -> obuf (bf16, heads concat)
  attn_kernel<<<dim3(BB * HH, TT / 128), 256, 0, stream>>>(qkv, vtbuf, obuf);

  // proj: x2 = x + obuf @ wproj + bproj  (f32 out)
  gemm_bf16_kernel<false, true, false, false, true>
      <<<dim3(NTOK / 128, EE / 128), 256, 0, stream>>>(
      obuf, wpj_t, bproj, x, nullptr, nullptr, x2, NTOK, EE, EE);

  // LN2 -> hbuf (bf16)
  ln_kernel<<<NTOK, 256, 0, stream>>>(x2, ln2_g, ln2_b, hbuf);

  // FC + exact GELU: m1 = gelu(hbuf @ w_fc + b_fc)  (bf16 out)
  gemm_bf16_kernel<true, false, true, false, false>
      <<<dim3(NTOK / 128, E4 / 128), 256, 0, stream>>>(
      hbuf, wfc_t, b_fc, nullptr, m1, nullptr, nullptr, NTOK, E4, EE);

  // out: d_out = x2 + m1 @ w_out + b_out  (f32 out)
  gemm_bf16_kernel<false, true, false, false, true>
      <<<dim3(NTOK / 128, EE / 128), 256, 0, stream>>>(
      m1, wo_t, b_out, x2, nullptr, nullptr, out, NTOK, EE, E4);
}
